// GraphSAGE_63359357550605
// MI455X (gfx1250) — matrix operand order repaired and checked by reading, not compiled, not run on hardware
//
#include <hip/hip_runtime.h>
#include <math.h>

// ---------------------------------------------------------------------------
// GraphSAGE (3-layer, mean aggregation) for MI455X / gfx1250, wave32 + WMMA.
//   h = relu(sage(x)); h = relu(sage(h)); out = log_softmax(sage(h))
//   sage(h) = (segment_mean over edges)(h) @ Wl^T + h @ Wr^T + b
//
// fp32-accurate GEMM via bf16 hi/lo split: x ~= hi + lo, keep
// hi*hi + hi*lo + lo*hi (3 WMMAs). Weights pre-split once into packed bf16
// hi/lo buffers so the hot loop has no B-side conversion VALU.
// ---------------------------------------------------------------------------

typedef __attribute__((ext_vector_type(16))) __bf16 v16bf;
typedef __attribute__((ext_vector_type(8)))  float  v8f;

#define DK   128        // inner dimension (D_IN == D_HID == 128)
#define KW   (DK / 2)   // packed (2 x bf16) words per weight row

// ---- bf16 round-to-nearest-even helpers -----------------------------------
__device__ __forceinline__ unsigned int bf16_rne(float f) {
    unsigned int u = __float_as_uint(f);
    return (u + 0x7FFFu + ((u >> 16) & 1u)) >> 16;   // 16-bit result in low bits
}
__device__ __forceinline__ float bf16_to_f32(unsigned int h) {
    return __uint_as_float(h << 16);
}

struct FragPair { v16bf hi; v16bf lo; };
union FragBits { v16bf v; unsigned int u[8]; };

// Split a float pair into packed (hi, lo) bf16 words: x ~= hi + lo.
__device__ __forceinline__ void split_pack(float x0, float x1,
                                           unsigned int& hw, unsigned int& lw) {
    unsigned int h0 = bf16_rne(x0);
    unsigned int h1 = bf16_rne(x1);
    unsigned int l0 = bf16_rne(x0 - bf16_to_f32(h0));
    unsigned int l1 = bf16_rne(x1 - bf16_to_f32(h1));
    hw = (h1 << 16) | h0;
    lw = (l1 << 16) | l0;
}

// A-fragment (16x32 bf16, this lane's slice): elements
//   u[0..3] <- row[k0 + 16*half + 0..7],  u[4..7] <- row[k0 + 16*half + 8..15]
__device__ __forceinline__ FragPair make_a_frag(const float* __restrict__ rowp,
                                                int k0, int half, float scale) {
    FragBits H, L;
    const float* p0 = rowp + k0 + 16 * half;
    const float* p1 = p0 + 8;
#pragma unroll
    for (int i = 0; i < 4; ++i) {
        split_pack(p0[2 * i] * scale, p0[2 * i + 1] * scale, H.u[i],     L.u[i]);
        split_pack(p1[2 * i] * scale, p1[2 * i + 1] * scale, H.u[4 + i], L.u[4 + i]);
    }
    FragPair fp; fp.hi = H.v; fp.lo = L.v; return fp;
}

// B-fragment from pre-packed bf16 words. Word j of row n = bf16(W[n][2j]),
// bf16(W[n][2j+1]) — the same contiguous K order make_a_frag uses:
// lane holds K = k0 + 16*half + {2v, 2v+1}  ->  words (k0+16*half)/2 + v.
__device__ __forceinline__ v16bf load_b_packed(const unsigned int* __restrict__ wrow,
                                               int k0, int half) {
    FragBits f;
    const unsigned int* p = wrow + ((k0 + 16 * half) >> 1);
#pragma unroll
    for (int i = 0; i < 8; ++i) f.u[i] = p[i];   // two b128 loads
    return f.v;
}

#define WMMA_BF16(A, B, C) \
    __builtin_amdgcn_wmma_f32_16x16x32_bf16(false, (A), false, (B), (short)0, (C), false, false)

// ---------------------------------------------------------------------------
// One-time weight split: W [rows x 128] f32 -> packed hi/lo bf16 word arrays
// [rows x 64] (pairs contiguous along K, matching load_b_packed).
// ---------------------------------------------------------------------------
__global__ void pack_w_k(const float* __restrict__ W,
                         unsigned int* __restrict__ hi,
                         unsigned int* __restrict__ lo, int n_pairs) {
    int i = blockIdx.x * blockDim.x + threadIdx.x;
    if (i >= n_pairs) return;
    unsigned int hw, lw;
    split_pack(W[2 * i], W[2 * i + 1], hw, lw);
    hi[i] = hw;
    lo[i] = lw;
}

// ---------------------------------------------------------------------------
// Fused SAGE linear:  out = relu?((agg .* inv_deg) @ Wl^T + h @ Wr^T + b)
// grid.x = ceil(N/16) row tiles; blockDim = (32, n_col_tiles); one wave per
// 16x16 output tile; 24 v_wmma per tile. EXEC is all-ones at every WMMA
// (masking only in the store epilogue).
// ---------------------------------------------------------------------------
__global__ void sage_gemm_wmma(const float* __restrict__ agg,
                               const float* __restrict__ h,
                               const float* __restrict__ inv_deg,
                               const unsigned int* __restrict__ WlHi,
                               const unsigned int* __restrict__ WlLo,
                               const unsigned int* __restrict__ WrHi,
                               const unsigned int* __restrict__ WrLo,
                               const float* __restrict__ bias,
                               float* __restrict__ out,
                               int n_rows, int d_out, int do_relu) {
    const int lane = threadIdx.x;        // 0..31
    const int half = lane >> 4;          // 0 | 1
    const int lrow = lane & 15;
    const int m0   = blockIdx.x * 16;
    const int n0   = threadIdx.y * 16;

    int arow = m0 + lrow; if (arow > n_rows - 1) arow = n_rows - 1;
    int ncol = n0 + lrow; if (ncol > d_out - 1)  ncol = d_out - 1;

    const float* aggp = agg + (long long)arow * DK;
    const float* hp   = h   + (long long)arow * DK;
    const unsigned int* wlh = WlHi + (long long)ncol * KW;
    const unsigned int* wll = WlLo + (long long)ncol * KW;
    const unsigned int* wrh = WrHi + (long long)ncol * KW;
    const unsigned int* wrl = WrLo + (long long)ncol * KW;
    const float scale = inv_deg[arow];

    v8f c = {};
#pragma unroll
    for (int k0 = 0; k0 < DK; k0 += 32) {
        // path 1: mean-aggregated neighbors through Wl
        FragPair a  = make_a_frag(aggp, k0, half, scale);
        v16bf    bh = load_b_packed(wlh, k0, half);
        v16bf    bl = load_b_packed(wll, k0, half);
        c = WMMA_BF16(a.hi, bh, c);
        c = WMMA_BF16(a.hi, bl, c);
        c = WMMA_BF16(a.lo, bh, c);
        // path 2: self features through Wr
        a  = make_a_frag(hp, k0, half, 1.0f);
        bh = load_b_packed(wrh, k0, half);
        bl = load_b_packed(wrl, k0, half);
        c = WMMA_BF16(a.hi, bh, c);
        c = WMMA_BF16(a.hi, bl, c);
        c = WMMA_BF16(a.lo, bh, c);
    }

    // C/D layout: VGPR r -> row m0 + r + 8*half, col n0 + (lane & 15)
    const int  col   = n0 + lrow;
    const bool colok = col < d_out;
    const float bv   = colok ? bias[col] : 0.0f;
#pragma unroll
    for (int r = 0; r < 8; ++r) {
        int   row = m0 + r + 8 * half;
        float v   = c[r] + bv;
        if (do_relu) v = fmaxf(v, 0.0f);
        if (colok && row < n_rows) out[(long long)row * d_out + col] = v;
    }
}

// ---------------------------------------------------------------------------
// Edge scatter: one wave per edge, lane handles 4 contiguous floats (float4
// gather, 4 x global_atomic_add_f32 scatter). x and agg are L2-resident
// (51 MB each vs 192 MB L2), so this is L2-atomic-bandwidth bound.
// ---------------------------------------------------------------------------
__global__ void scatter_add_k(const float* __restrict__ h,
                              const long long* __restrict__ src,
                              const long long* __restrict__ dst,
                              float* __restrict__ agg, int nE) {
    int t = blockIdx.x * blockDim.x + threadIdx.x;
    int e = t >> 5;
    int lane = t & 31;
    if (e >= nE) return;
    long long s = src[e];
    long long d = dst[e];
    const float4* xp = (const float4*)(h + s * DK);
    float4 v = xp[lane];
    float* ap = agg + d * DK + lane * 4;
    atomicAdd(ap + 0, v.x);
    atomicAdd(ap + 1, v.y);
    atomicAdd(ap + 2, v.z);
    atomicAdd(ap + 3, v.w);
}

__global__ void deg_count_k(const long long* __restrict__ dst,
                            float* __restrict__ deg, int nE) {
    int e = blockIdx.x * blockDim.x + threadIdx.x;
    if (e < nE) atomicAdd(&deg[dst[e]], 1.0f);
}

__global__ void inv_deg_k(const float* __restrict__ deg,
                          float* __restrict__ inv, int n) {
    int i = blockIdx.x * blockDim.x + threadIdx.x;
    if (i < n) inv[i] = 1.0f / fmaxf(deg[i], 1.0f);
}

// ---------------------------------------------------------------------------
// In-place log_softmax over 40 columns; one wave per row (wave32 shuffles).
// Lane owns cols {lane, lane+32 (lanes 0..7)}.
// ---------------------------------------------------------------------------
__global__ void log_softmax40_k(float* __restrict__ out, int n_rows) {
    int t = blockIdx.x * blockDim.x + threadIdx.x;
    int row = t >> 5;
    int lane = t & 31;
    if (row >= n_rows) return;
    float* p = out + (long long)row * 40;
    float v0 = p[lane];
    float v1 = (lane < 8) ? p[lane + 32] : -__builtin_inff();
    float m = fmaxf(v0, v1);
#pragma unroll
    for (int o = 16; o > 0; o >>= 1) m = fmaxf(m, __shfl_xor(m, o, 32));
    float s = expf(v0 - m) + ((lane < 8) ? expf(v1 - m) : 0.0f);
#pragma unroll
    for (int o = 16; o > 0; o >>= 1) s += __shfl_xor(s, o, 32);
    float l = logf(s);
    p[lane] = v0 - m - l;
    if (lane < 8) p[lane + 32] = v1 - m - l;
}

// ---------------------------------------------------------------------------
extern "C" void kernel_launch(void* const* d_in, const int* in_sizes, int n_in,
                              void* d_out, int out_size, void* d_ws, size_t ws_size,
                              hipStream_t stream) {
    const float*     x   = (const float*)d_in[0];
    const long long* ei  = (const long long*)d_in[1];   // int64 [2, E]
    const float*     Wl0 = (const float*)d_in[2];
    const float*     Wr0 = (const float*)d_in[3];
    const float*     b0  = (const float*)d_in[4];
    const float*     Wl1 = (const float*)d_in[5];
    const float*     Wr1 = (const float*)d_in[6];
    const float*     b1  = (const float*)d_in[7];
    const float*     Wl2 = (const float*)d_in[8];
    const float*     Wr2 = (const float*)d_in[9];
    const float*     b2  = (const float*)d_in[10];

    const int N    = in_sizes[0] / DK;     // 100000
    const int E    = in_sizes[1] / 2;      // 800000
    const int DOUT = in_sizes[10];         // 40
    const long long* src = ei;
    const long long* dst = ei + E;

    // workspace carve-out: agg | h1 | h2 | deg | inv | packed weights
    float* agg = (float*)d_ws;
    float* h1  = agg + (size_t)N * DK;
    float* h2  = h1  + (size_t)N * DK;
    float* deg = h2  + (size_t)N * DK;
    float* inv = deg + (size_t)N;
    unsigned int* pw = (unsigned int*)(inv + (size_t)N);
    const size_t wsz  = (size_t)DK * KW;      // packed words per 128x128 matrix
    const size_t wsz2 = (size_t)DOUT * KW;    // packed words per 40x128 matrix
    unsigned int* wl0h = pw;            unsigned int* wl0l = wl0h + wsz;
    unsigned int* wr0h = wl0l + wsz;    unsigned int* wr0l = wr0h + wsz;
    unsigned int* wl1h = wr0l + wsz;    unsigned int* wl1l = wl1h + wsz;
    unsigned int* wr1h = wl1l + wsz;    unsigned int* wr1l = wr1h + wsz;
    unsigned int* wl2h = wr1l + wsz;    unsigned int* wl2l = wl2h + wsz2;
    unsigned int* wr2h = wl2l + wsz2;   unsigned int* wr2l = wr2h + wsz2;
    (void)ws_size; (void)n_in; (void)out_size;

    const size_t featBytes   = (size_t)N * DK * sizeof(float);
    const int scatterBlocks  = (int)(((size_t)E * 32 + 255) / 256);
    const int mTiles         = (N + 15) / 16;
    const int pairs  = (int)wsz;   // 8192
    const int pairs2 = (int)wsz2;  // 2560

    // one-time weight hi/lo split (tiny)
    pack_w_k<<<(pairs  + 255) / 256, 256, 0, stream>>>(Wl0, wl0h, wl0l, pairs);
    pack_w_k<<<(pairs  + 255) / 256, 256, 0, stream>>>(Wr0, wr0h, wr0l, pairs);
    pack_w_k<<<(pairs  + 255) / 256, 256, 0, stream>>>(Wl1, wl1h, wl1l, pairs);
    pack_w_k<<<(pairs  + 255) / 256, 256, 0, stream>>>(Wr1, wr1h, wr1l, pairs);
    pack_w_k<<<(pairs2 + 255) / 256, 256, 0, stream>>>(Wl2, wl2h, wl2l, pairs2);
    pack_w_k<<<(pairs2 + 255) / 256, 256, 0, stream>>>(Wr2, wr2h, wr2l, pairs2);

    // degree (shared by all three layers)
    hipMemsetAsync(deg, 0, (size_t)N * sizeof(float), stream);
    deg_count_k<<<(E + 255) / 256, 256, 0, stream>>>(dst, deg, E);
    inv_deg_k<<<(N + 255) / 256, 256, 0, stream>>>(deg, inv, N);

    // ---- layer 0: x -> h1 (relu) ----
    hipMemsetAsync(agg, 0, featBytes, stream);
    scatter_add_k<<<scatterBlocks, 256, 0, stream>>>(x, src, dst, agg, E);
    sage_gemm_wmma<<<dim3(mTiles), dim3(32, DK / 16), 0, stream>>>(
        agg, x, inv, wl0h, wl0l, wr0h, wr0l, b0, h1, N, DK, 1);

    // ---- layer 1: h1 -> h2 (relu) ----
    hipMemsetAsync(agg, 0, featBytes, stream);
    scatter_add_k<<<scatterBlocks, 256, 0, stream>>>(h1, src, dst, agg, E);
    sage_gemm_wmma<<<dim3(mTiles), dim3(32, DK / 16), 0, stream>>>(
        agg, h1, inv, wl1h, wl1l, wr1h, wr1l, b1, h2, N, DK, 1);

    // ---- layer 2: h2 -> d_out (no relu), then log_softmax in place ----
    hipMemsetAsync(agg, 0, featBytes, stream);
    scatter_add_k<<<scatterBlocks, 256, 0, stream>>>(h2, src, dst, agg, E);
    const int nTiles2 = (DOUT + 15) / 16;   // 3 tiles, col<40 masked at store
    sage_gemm_wmma<<<dim3(mTiles), dim3(32, nTiles2), 0, stream>>>(
        agg, h2, inv, wl2h, wl2l, wr2h, wr2l, b2, (float*)d_out, N, DOUT, 0);

    log_softmax40_k<<<(int)(((size_t)N * 32 + 255) / 256), 256, 0, stream>>>(
        (float*)d_out, N);
}
